// MultiHeadAttention2D_10290741641457
// MI455X (gfx1250) — matrix operand order repaired and checked by reading, not compiled, not run on hardware
//
#include <hip/hip_runtime.h>

// ---------------------------------------------------------------------------
// Problem constants (from reference): B=2, D=1024, S=2048, H=16, HD=64
// ---------------------------------------------------------------------------
static constexpr int B  = 2;
static constexpr int D  = 1024;
static constexpr int S  = 2048;
static constexpr int H  = 16;
static constexpr int HD = 64;
static constexpr int BH = B * H;
static constexpr float SM_SCALE = 0.125f;   // 1/sqrt(64)

typedef __attribute__((ext_vector_type(16))) __bf16 v16bf;
typedef __attribute__((ext_vector_type(8)))  __bf16 v8bf;
typedef __attribute__((ext_vector_type(8)))  float  v8f;

// ---------------------------------------------------------------------------
// WMMA helper: D = A(16x32 bf16) * B(32x16 bf16) + C(16x16 f32)
// ---------------------------------------------------------------------------
__device__ __forceinline__ v8f wmma_bf16(v16bf a, v16bf b, v8f c) {
  return __builtin_amdgcn_wmma_f32_16x16x32_bf16(
      /*neg_a=*/false, a, /*neg_b=*/false, b,
      /*c_mod=*/(short)0, c, /*reuse_a=*/false, /*reuse_b=*/false);
}

// Build a 16-element fragment from two 16-byte (8 x bf16) aligned chunks.
__device__ __forceinline__ v16bf load_frag2x16(const __bf16* p0, const __bf16* p1) {
  v8bf lo = *(const v8bf*)p0;
  v8bf hi = *(const v8bf*)p1;
  return __builtin_shufflevector(lo, hi, 0, 1, 2, 3, 4, 5, 6, 7,
                                 8, 9, 10, 11, 12, 13, 14, 15);
}

// A-fragment (16x32 bf16), row-major source with row stride ld (elements).
// Contiguous-K layout used here: lane (l,hi): row M=l, elems[i] = K = hi*16 + i.
__device__ __forceinline__ v16bf load_a_frag(const __bf16* base, int ld, int l, int hi) {
  const __bf16* p = base + l * ld + hi * 16;
  return load_frag2x16(p, p + 8);
}

// B-fragment (32x16 bf16) with element (k,n) at base[n*ld + k]
// (K contiguous per column). Contiguous-K layout used here: lane n=l, elems[i] = K = hi*16 + i.
__device__ __forceinline__ v16bf load_b_frag_kcontig(const __bf16* base, int ld, int l, int hi) {
  const __bf16* p = base + l * ld + hi * 16;
  return load_frag2x16(p, p + 8);
}

// bf16 pair -> u32 packing for LDS staging stores.
union PackBF2 { __bf16 h[2]; unsigned int u; };

// gfx1250 async global->LDS 16B copy (ASYNCcnt-tracked, no VGPR data path).
__device__ __forceinline__ void async_copy_b128(unsigned int lds_off, const void* gptr) {
  asm volatile("global_load_async_to_lds_b128 %0, %1, off"
               :: "v"(lds_off), "v"(gptr) : "memory");
}
__device__ __forceinline__ void wait_async0() {
  asm volatile("s_wait_asynccnt 0x0" ::: "memory");
}
__device__ __forceinline__ unsigned int lds_offset(const void* p) {
  return (unsigned int)(unsigned long long)p;   // low 32 bits = LDS offset
}

// ---------------------------------------------------------------------------
// Kernel 0: one-time f32 -> bf16 conversion of the four weight matrices so
// the GEMM hot loops can stage B tiles with pure async copies.
// ---------------------------------------------------------------------------
__global__ __launch_bounds__(256) void cvt_w_kernel(
    const float* __restrict__ w0, const float* __restrict__ w1,
    const float* __restrict__ w2, const float* __restrict__ w3,
    __bf16* __restrict__ o0, __bf16* __restrict__ o1,
    __bf16* __restrict__ o2, __bf16* __restrict__ o3) {
  const int which = blockIdx.y;
  const float* src = (which == 0) ? w0 : (which == 1) ? w1 : (which == 2) ? w2 : w3;
  __bf16* dst      = (which == 0) ? o0 : (which == 1) ? o1 : (which == 2) ? o2 : o3;
  const int i = (blockIdx.x * 256 + threadIdx.x) * 4;
  const float4 f = *(const float4*)(src + i);
  PackBF2 a, c;
  a.h[0] = (__bf16)f.x; a.h[1] = (__bf16)f.y;
  c.h[0] = (__bf16)f.z; c.h[1] = (__bf16)f.w;
  *(unsigned int*)(dst + i)     = a.u;
  *(unsigned int*)(dst + i + 2) = c.u;
}

// ---------------------------------------------------------------------------
// Kernel 1: fused QKV projection, double-buffered single-barrier pipeline.
//   Y[b,s,e] = sum_d x[b,d,s] * W[e,d] + bias[e]
// Q,K written [BH][S][HD]; V written transposed [BH][HD][S] so attention can
// stage V^T panels with async copies (and V's epilogue becomes 16B stores).
// ---------------------------------------------------------------------------
__global__ __launch_bounds__(256) void qkv_proj_kernel(
    const float* __restrict__ x,
    const __bf16* __restrict__ wqb, const float* __restrict__ bq,
    const __bf16* __restrict__ wkb, const float* __restrict__ bk,
    const __bf16* __restrict__ wvb, const float* __restrict__ bv,
    __bf16* __restrict__ qo, __bf16* __restrict__ ko, __bf16* __restrict__ vo) {
  constexpr int BM = 128, BN = 128, BK = 32;
  constexpr int LDA = BK + 8;   // 40 elems = 80B rows (16B-aligned)
  constexpr int LDK = BK + 8;   // sB column-major [n][k]
  constexpr int KT  = D / BK;   // 32 k-steps
  __shared__ __bf16 sA[2][BM * LDA];
  __shared__ __bf16 sB[2][BN * LDK];

  const int which = blockIdx.z;
  const __bf16* w   = (which == 0) ? wqb : (which == 1) ? wkb : wvb;
  const float* bias = (which == 0) ? bq  : (which == 1) ? bk  : bv;
  __bf16* dst       = (which == 0) ? qo  : (which == 1) ? ko  : vo;

  const int n0 = blockIdx.x * BN;
  const int g0 = blockIdx.y * BM;     // global row = b*S + s
  const int b  = g0 / S;
  const int s0 = g0 % S;

  const int tid  = threadIdx.x;
  const int wid  = tid >> 5;
  const int lane = tid & 31;
  const int l    = lane & 15;
  const int hi   = lane >> 4;
  const int wm   = wid & 3;           // 4 waves along M (32 rows each)
  const int wn   = wid >> 2;          // 2 waves along N (64 cols each)

  const float* xb = x + (size_t)b * D * S;

  // Staging helpers -------------------------------------------------------
  const int a_mm = tid & 127;
  const int a_k2 = (tid >> 7) * 2;
  const int b_nn = tid >> 1;
  const int b_k0 = (tid & 1) * 16;

  auto stageA = [&](int buf, int kt) {   // transpose+convert from x (VGPR path)
    const int k0 = kt * BK;
#pragma unroll
    for (int pp = 0; pp < 8; ++pp) {
      const int kk = pp * 4 + a_k2;
      PackBF2 cv;
      cv.h[0] = (__bf16)xb[(size_t)(k0 + kk)     * S + (s0 + a_mm)];
      cv.h[1] = (__bf16)xb[(size_t)(k0 + kk + 1) * S + (s0 + a_mm)];
      *(unsigned int*)&sA[buf][a_mm * LDA + kk] = cv.u;
    }
  };
  auto stageB = [&](int buf, int kt) {   // pure async copy from bf16 weights
    const __bf16* g = w + (size_t)(n0 + b_nn) * D + kt * BK + b_k0;
    const unsigned int loff = lds_offset(&sB[buf][b_nn * LDK + b_k0]);
    async_copy_b128(loff,      g);
    async_copy_b128(loff + 16, g + 8);
  };

  const v8f vzero = {};
  v8f acc[2][4];
#pragma unroll
  for (int i = 0; i < 2; ++i)
#pragma unroll
    for (int j = 0; j < 4; ++j) acc[i][j] = vzero;

  stageA(0, 0);
  stageB(0, 0);

  for (int kt = 0; kt < KT; ++kt) {
    const int cur = kt & 1;
    wait_async0();
    __syncthreads();                     // flushes DS stores + syncs asyncs
    if (kt + 1 < KT) { stageA(cur ^ 1, kt + 1); stageB(cur ^ 1, kt + 1); }

    v16bf af[2], bfr[4];
#pragma unroll
    for (int mf = 0; mf < 2; ++mf)
      af[mf] = load_a_frag(&sA[cur][(wm * 32 + mf * 16) * LDA], LDA, l, hi);
#pragma unroll
    for (int nf = 0; nf < 4; ++nf)
      bfr[nf] = load_b_frag_kcontig(&sB[cur][(wn * 64 + nf * 16) * LDK], LDK, l, hi);
#pragma unroll
    for (int mf = 0; mf < 2; ++mf)
#pragma unroll
      for (int nf = 0; nf < 4; ++nf)
        acc[mf][nf] = wmma_bf16(af[mf], bfr[nf], acc[mf][nf]);
  }

  // Epilogue: add bias; Q/K scatter [BH][S][HD]; V packed 16B rows [BH][HD][S].
#pragma unroll
  for (int nf = 0; nf < 4; ++nf) {
    const int e  = n0 + wn * 64 + nf * 16 + l;
    const float be = bias[e];
    const int h  = e >> 6;
    const int hd = e & 63;
#pragma unroll
    for (int mf = 0; mf < 2; ++mf) {
      const int sbase = s0 + wm * 32 + mf * 16 + hi * 8;
      if (which == 2) {
        v8bf pk;
#pragma unroll
        for (int v = 0; v < 8; ++v) pk[v] = (__bf16)(acc[mf][nf][v] + be);
        *(v8bf*)&dst[((size_t)(b * H + h) * HD + hd) * S + sbase] = pk;
      } else {
#pragma unroll
        for (int v = 0; v < 8; ++v)
          dst[(((size_t)(b * H + h)) * S + (sbase + v)) * HD + hd] =
              (__bf16)(acc[mf][nf][v] + be);
      }
    }
  }
}

// ---------------------------------------------------------------------------
// Kernel 2: flash attention per (b,h), double-buffered async K / V^T panels.
// Block = 256 threads = 8 waves; each wave owns 16 query rows.
// ---------------------------------------------------------------------------
__global__ __launch_bounds__(256) void attn_kernel(
    const __bf16* __restrict__ qws, const __bf16* __restrict__ kws,
    const __bf16* __restrict__ vtws, __bf16* __restrict__ ctx) {
  constexpr int LDT  = HD + 8;   // 72 elems = 144B rows (16B-aligned)
  constexpr int LDVT = 64 + 8;   // sVT rows: 64 keys + pad
  constexpr int NKB  = S / 64;   // 32 key panels
  __shared__ __bf16 sK [2][64 * LDT];    // [key][d]
  __shared__ __bf16 sVT[2][HD * LDVT];   // [d][key]
  __shared__ __bf16 sP [8 * 16 * LDT];

  const int bh = blockIdx.y;
  const int b  = bh / H;
  const int h  = bh % H;
  const int tid  = threadIdx.x;
  const int wid  = tid >> 5;
  const int lane = tid & 31;
  const int l    = lane & 15;
  const int hi   = lane >> 4;
  const int q0   = blockIdx.x * 128 + wid * 16;

  const __bf16* Qg  = qws  + (size_t)bh * S * HD;   // [s][d]
  const __bf16* Kg  = kws  + (size_t)bh * S * HD;   // [s][d]
  const __bf16* Vtg = vtws + (size_t)bh * HD * S;   // [d][s]

  // Async panel staging: both K and V^T are pure 2D copies.
  const int st_r = tid >> 2;            // 0..63
  const int st_c = (tid & 3) * 16;      // 0,16,32,48
  auto stage = [&](int buf, int kb) {
    {
      const unsigned int loff = lds_offset(&sK[buf][st_r * LDT + st_c]);
      const __bf16* g = Kg + (size_t)(kb + st_r) * HD + st_c;
      async_copy_b128(loff,      g);
      async_copy_b128(loff + 16, g + 8);
    }
    {
      const unsigned int loff = lds_offset(&sVT[buf][st_r * LDVT + st_c]);
      const __bf16* g = Vtg + (size_t)st_r * S + kb + st_c;
      async_copy_b128(loff,      g);
      async_copy_b128(loff + 16, g + 8);
    }
  };

  // Q fragments live in registers for the whole K loop.
  v16bf aq[2];
#pragma unroll
  for (int dk = 0; dk < 2; ++dk)
    aq[dk] = load_a_frag(Qg + (size_t)q0 * HD + dk * 32, HD, l, hi);

  const v8f vzero = {};
  v8f o[4] = {vzero, vzero, vzero, vzero};   // O tile 16x64 (4 C-tiles)
  float mrow[8], lrow[8];                    // per-row stats, row = v + hi*8
#pragma unroll
  for (int v = 0; v < 8; ++v) { mrow[v] = -3.0e38f; lrow[v] = 0.0f; }

  __bf16* sPw = sP + wid * 16 * LDT;         // private P region per wave

  stage(0, 0);

  for (int it = 0; it < NKB; ++it) {
    const int cur = it & 1;
    wait_async0();
    __syncthreads();
    if (it + 1 < NKB) stage(cur ^ 1, (it + 1) * 64);

    // Scores S = Q * K^T : 4 N-tiles of 16 keys; accumulate d in 2 steps.
    v8f sc[4];
#pragma unroll
    for (int nt = 0; nt < 4; ++nt) {
      v8f a = vzero;
#pragma unroll
      for (int dk = 0; dk < 2; ++dk) {
        const __bf16* p = &sK[cur][(nt * 16 + l) * LDT + dk * 32 + hi * 16];
        a = wmma_bf16(aq[dk], load_frag2x16(p, p + 8), a);
      }
      sc[nt] = a * SM_SCALE;
    }

    // Online softmax. C layout: row = v + hi*8 (elem v); col = nt*16 + l.
    float scale[8], rs[8];
#pragma unroll
    for (int v = 0; v < 8; ++v) {
      float mx = fmaxf(fmaxf(sc[0][v], sc[1][v]), fmaxf(sc[2][v], sc[3][v]));
#pragma unroll
      for (int msk = 1; msk < 16; msk <<= 1)
        mx = fmaxf(mx, __shfl_xor(mx, msk, 32));
      const float mnew = fmaxf(mrow[v], mx);
      scale[v] = __expf(mrow[v] - mnew);
      mrow[v]  = mnew;
      rs[v]    = 0.0f;
    }
#pragma unroll
    for (int nt = 0; nt < 4; ++nt)
#pragma unroll
      for (int v = 0; v < 8; ++v) {
        const float p = __expf(sc[nt][v] - mrow[v]);
        sc[nt][v] = p;
        rs[v] += p;
      }
#pragma unroll
    for (int v = 0; v < 8; ++v) {
      float r = rs[v];
#pragma unroll
      for (int msk = 1; msk < 16; msk <<= 1) r += __shfl_xor(r, msk, 32);
      lrow[v] = lrow[v] * scale[v] + r;
    }
#pragma unroll
    for (int nt = 0; nt < 4; ++nt)
#pragma unroll
      for (int v = 0; v < 8; ++v) o[nt][v] *= scale[v];

    // Re-layout P (C-format f32) -> A-format bf16 via this wave's LDS slice.
#pragma unroll
    for (int nt = 0; nt < 4; ++nt)
#pragma unroll
      for (int v = 0; v < 8; ++v)
        sPw[(v + hi * 8) * LDT + nt * 16 + l] = (__bf16)sc[nt][v];
    asm volatile("s_wait_dscnt 0x0" ::: "memory");  // wave-local LDS RAW fence

    // O += P(16x64) * V(64x64); V^T layout makes B-frags contiguous.
    v16bf ap[2];
#pragma unroll
    for (int kk = 0; kk < 2; ++kk)
      ap[kk] = load_a_frag(sPw + kk * 32, LDT, l, hi);
#pragma unroll
    for (int nt = 0; nt < 4; ++nt)
#pragma unroll
      for (int kk = 0; kk < 2; ++kk) {
        const __bf16* p = &sVT[cur][(nt * 16 + l) * LDVT + kk * 32 + hi * 16];
        o[nt] = wmma_bf16(ap[kk], load_frag2x16(p, p + 8), o[nt]);
      }
  }

  // Finalize: O /= l, store ctx as bf16 [B*S][D] with e = h*64 + d.
#pragma unroll
  for (int v = 0; v < 8; ++v) {
    const float inv = 1.0f / lrow[v];
    const int s = q0 + v + hi * 8;
#pragma unroll
    for (int nt = 0; nt < 4; ++nt)
      ctx[((size_t)(b * S + s)) * D + h * HD + nt * 16 + l] =
          (__bf16)(o[nt][v] * inv);
  }
}

// ---------------------------------------------------------------------------
// Kernel 3: output projection, fully-async double-buffered pipeline.
//   out[b,e,s] = sum_d ctx[b,s,d] * Wo[e,d] + bo[e]
// ---------------------------------------------------------------------------
__global__ __launch_bounds__(256) void out_proj_kernel(
    const __bf16* __restrict__ ctx, const __bf16* __restrict__ wob,
    const float* __restrict__ bo, float* __restrict__ out) {
  constexpr int BM = 128, BN = 128, BK = 32;
  constexpr int LDA = BK + 8;
  constexpr int LDK = BK + 8;
  constexpr int KT  = D / BK;
  __shared__ __bf16 sA[2][BM * LDA];
  __shared__ __bf16 sB[2][BN * LDK];

  const int n0 = blockIdx.x * BN;
  const int g0 = blockIdx.y * BM;
  const int tid  = threadIdx.x;
  const int wid  = tid >> 5;
  const int lane = tid & 31;
  const int l    = lane & 15;
  const int hi   = lane >> 4;
  const int wm   = wid & 3;
  const int wn   = wid >> 2;

  const int mm  = tid >> 1;
  const int kk0 = (tid & 1) * 16;
  auto stage = [&](int buf, int kt) {
    const int k0 = kt * BK;
    {
      const unsigned int loff = lds_offset(&sA[buf][mm * LDA + kk0]);
      const __bf16* g = ctx + (size_t)(g0 + mm) * D + k0 + kk0;
      async_copy_b128(loff,      g);
      async_copy_b128(loff + 16, g + 8);
    }
    {
      const unsigned int loff = lds_offset(&sB[buf][mm * LDK + kk0]);
      const __bf16* g = wob + (size_t)(n0 + mm) * D + k0 + kk0;
      async_copy_b128(loff,      g);
      async_copy_b128(loff + 16, g + 8);
    }
  };

  const v8f vzero = {};
  v8f acc[2][4];
#pragma unroll
  for (int i = 0; i < 2; ++i)
#pragma unroll
    for (int j = 0; j < 4; ++j) acc[i][j] = vzero;

  stage(0, 0);

  for (int kt = 0; kt < KT; ++kt) {
    const int cur = kt & 1;
    wait_async0();
    __syncthreads();
    if (kt + 1 < KT) stage(cur ^ 1, kt + 1);

    v16bf af[2], bfr[4];
#pragma unroll
    for (int mf = 0; mf < 2; ++mf)
      af[mf] = load_a_frag(&sA[cur][(wm * 32 + mf * 16) * LDA], LDA, l, hi);
#pragma unroll
    for (int nf = 0; nf < 4; ++nf)
      bfr[nf] = load_b_frag_kcontig(&sB[cur][(wn * 64 + nf * 16) * LDK], LDK, l, hi);
#pragma unroll
    for (int mf = 0; mf < 2; ++mf)
#pragma unroll
      for (int nf = 0; nf < 4; ++nf)
        acc[mf][nf] = wmma_bf16(af[mf], bfr[nf], acc[mf][nf]);
  }

  // Epilogue: out[b*D*S + e*S + s]; per lane the 8 v-rows are consecutive s.
#pragma unroll
  for (int nf = 0; nf < 4; ++nf) {
    const int e = n0 + wn * 64 + nf * 16 + l;
    const float be = bo[e];
#pragma unroll
    for (int mf = 0; mf < 2; ++mf) {
#pragma unroll
      for (int v = 0; v < 8; ++v) {
        const int g = g0 + wm * 32 + mf * 16 + v + hi * 8;
        const int bb = g >> 11;           // g / S
        const int ss = g & (S - 1);       // g % S
        out[(size_t)bb * D * S + (size_t)e * S + ss] = acc[mf][nf][v] + be;
      }
    }
  }
}

// ---------------------------------------------------------------------------
// Launch: cvt -> proj -> attention -> out-proj (stream-ordered).
// Workspace: Q,K [BH,S,HD], V^T [BH,HD,S], ctx [B*S,D] bf16 (8MB each)
//          + 4 bf16 weight copies (2MB each) = 40MB.
// ---------------------------------------------------------------------------
extern "C" void kernel_launch(void* const* d_in, const int* in_sizes, int n_in,
                              void* d_out, int out_size, void* d_ws, size_t ws_size,
                              hipStream_t stream) {
  (void)in_sizes; (void)n_in; (void)out_size; (void)ws_size;
  const float* x  = (const float*)d_in[0];
  const float* wq = (const float*)d_in[1];
  const float* bq = (const float*)d_in[2];
  const float* wk = (const float*)d_in[3];
  const float* bk = (const float*)d_in[4];
  const float* wv = (const float*)d_in[5];
  const float* bv = (const float*)d_in[6];
  const float* wo = (const float*)d_in[7];
  const float* bo = (const float*)d_in[8];
  float* out = (float*)d_out;

  const size_t nQKV = (size_t)BH * S * HD;   // 4,194,304 elems
  const size_t nW   = (size_t)D * D;         // 1,048,576 elems
  __bf16* qws  = (__bf16*)d_ws;
  __bf16* kws  = qws + nQKV;
  __bf16* vtws = kws + nQKV;                 // V stored [BH][HD][S]
  __bf16* ctx  = vtws + nQKV;
  __bf16* wqb  = ctx + nQKV;
  __bf16* wkb  = wqb + nW;
  __bf16* wvb  = wkb + nW;
  __bf16* wob  = wvb + nW;

  cvt_w_kernel<<<dim3(nW / 1024, 4), 256, 0, stream>>>(
      wq, wk, wv, wo, wqb, wkb, wvb, wob);
  qkv_proj_kernel<<<dim3(D / 128, (B * S) / 128, 3), 256, 0, stream>>>(
      x, wqb, bq, wkb, bk, wvb, bv, qws, kws, vtws);
  attn_kernel<<<dim3(S / 128, BH), 256, 0, stream>>>(qws, kws, vtws, ctx);
  out_proj_kernel<<<dim3(D / 128, (B * S) / 128), 256, 0, stream>>>(
      ctx, wob, bo, out);
}
